// LSTM_52304111730943
// MI455X (gfx1250) — compile-verified, matrix operand order checked by reading, not run on hardware
//
#include <hip/hip_runtime.h>

#define HIDDEN 64
#define FEAT   9
#define SEQ    128
#define BATCH  8192
#define NCLS   2
#define GATES  256   // 4*HIDDEN
#define MTILE  16

typedef _Float16 v16h __attribute__((ext_vector_type(16)));
typedef float    v8f  __attribute__((ext_vector_type(8)));

#define WMMA(A, B, C) __builtin_amdgcn_wmma_f32_16x16x32_f16( \
    false, (A), false, (B), (short)0, (C), false, false)

// A-operand gather from LDS (row-major [16][ld] f16) per the CDNA5 16-bit
// A-matrix 16x32 layout: lane m = lane&15, kb = (lane>>4)*8 (+32*kchunk),
// halves 0..7 -> K=kb..kb+7, halves 8..15 -> K=kb+16..kb+23.
static __device__ __forceinline__ v16h ldsA(const _Float16* base, int ld, int kc) {
  const int lane = threadIdx.x & 31;
  const int m  = lane & 15;
  const int kb = ((lane >> 4) << 3) + (kc << 5);
  const _Float16* p = base + m * ld + kb;
  v16h a;
#pragma unroll
  for (int i = 0; i < 8; ++i) { a[i] = p[i]; a[i + 8] = p[i + 16]; }
  return a;
}

// B-operand fragment built from fp32 weight W[n][k] (B = W^T, i.e. B[k][n]).
// Lane n = n0 + (lane&15); same kb split as A (mirrored M<->N).
static __device__ __forceinline__ v16h loadB(const float* W, int ld, int n0, int kc) {
  const int lane = threadIdx.x & 31;
  const int n  = n0 + (lane & 15);
  const int kb = ((lane >> 4) << 3) + (kc << 5);
  const float* p = W + n * ld + kb;
  v16h b;
#pragma unroll
  for (int i = 0; i < 8; ++i) { b[i] = (_Float16)p[i]; b[i + 8] = (_Float16)p[i + 16]; }
  return b;
}

// Same, but K zero-padded beyond kmax (for the FEAT=9 input projection).
static __device__ __forceinline__ v16h loadB_pad(const float* W, int ld, int kmax, int n0) {
  const int lane = threadIdx.x & 31;
  const int n  = n0 + (lane & 15);
  const int kb = (lane >> 4) << 3;
  v16h b;
#pragma unroll
  for (int i = 0; i < 8; ++i) {
    int k0 = kb + i, k1 = kb + 16 + i;
    b[i]     = (k0 < kmax) ? (_Float16)W[n * ld + k0] : (_Float16)0.0f;
    b[i + 8] = (k1 < kmax) ? (_Float16)W[n * ld + k1] : (_Float16)0.0f;
  }
  return b;
}

static __device__ __forceinline__ float fsig(float x)  { return 1.0f / (1.0f + __expf(-x)); }
static __device__ __forceinline__ float ftanh(float x) { return 2.0f / (1.0f + __expf(-2.0f * x)) - 1.0f; }

__global__ __launch_bounds__(256) void lstm2_fused_wmma(
    const float* __restrict__ x,
    const float* __restrict__ Wih0, const float* __restrict__ Whh0,
    const float* __restrict__ bih0, const float* __restrict__ bhh0,
    const float* __restrict__ Wih1, const float* __restrict__ Whh1,
    const float* __restrict__ bih1, const float* __restrict__ bhh1,
    const float* __restrict__ Wfc,  const float* __restrict__ bfc,
    float* __restrict__ out) {
  __shared__ _Float16 sh_x [MTILE * 32];       // x tile, K padded 9->32
  __shared__ _Float16 sh_h0[MTILE * HIDDEN];   // layer0 hidden state (f16)
  __shared__ _Float16 sh_h1[MTILE * HIDDEN];   // layer1 hidden state (f16)
  __shared__ float    sh_g [MTILE * GATES];    // gate pre-activations (fp32)
  __shared__ float    sh_b0[GATES];
  __shared__ float    sh_b1[GATES];

  const int tid   = threadIdx.x;
  const int wave  = tid >> 5;
  const int bbase = blockIdx.x * MTILE;

  // Fused biases + zero initial state.
  sh_b0[tid] = bih0[tid] + bhh0[tid];
  sh_b1[tid] = bih1[tid] + bhh1[tid];
  for (int i = tid; i < MTILE * HIDDEN; i += 256) {
    sh_h0[i] = (_Float16)0.0f;
    sh_h1[i] = (_Float16)0.0f;
  }

  // Resident weight fragments: 14 x v16h = 112 VGPRs/wave, reused 128 steps.
  const int n0 = wave * 32, n1 = wave * 32 + 16;
  const v16h Bih0a  = loadB_pad(Wih0, FEAT, FEAT, n0);
  const v16h Bih0b  = loadB_pad(Wih0, FEAT, FEAT, n1);
  const v16h Bhh0a0 = loadB(Whh0, HIDDEN, n0, 0), Bhh0a1 = loadB(Whh0, HIDDEN, n0, 1);
  const v16h Bhh0b0 = loadB(Whh0, HIDDEN, n1, 0), Bhh0b1 = loadB(Whh0, HIDDEN, n1, 1);
  const v16h Bih1a0 = loadB(Wih1, HIDDEN, n0, 0), Bih1a1 = loadB(Wih1, HIDDEN, n0, 1);
  const v16h Bih1b0 = loadB(Wih1, HIDDEN, n1, 0), Bih1b1 = loadB(Wih1, HIDDEN, n1, 1);
  const v16h Bhh1a0 = loadB(Whh1, HIDDEN, n0, 0), Bhh1a1 = loadB(Whh1, HIDDEN, n0, 1);
  const v16h Bhh1b0 = loadB(Whh1, HIDDEN, n1, 0), Bhh1b1 = loadB(Whh1, HIDDEN, n1, 1);

  // Cell state in registers: thread owns (m = p*4 + tid/64, n = tid&63).
  float c0[4] = {0.f, 0.f, 0.f, 0.f};
  float c1[4] = {0.f, 0.f, 0.f, 0.f};
  const int en  = tid & 63;
  const int emb = tid >> 6;

  // C/D-matrix scatter coords (32-bit 16x16 layout): lane -> N, VGPR r -> M.
  const int lane = tid & 31;
  const int cn   = lane & 15;
  const int cmb  = (lane >> 4) << 3;

  for (int t = 0; t < SEQ; ++t) {
    // Stage x[:, t, :] as f16, zero-padded to K=32.
#pragma unroll
    for (int i = 0; i < 2; ++i) {
      int idx = tid + i * 256;
      int m = idx >> 5, k = idx & 31;
      float v = (k < FEAT) ? x[((size_t)(bbase + m) * SEQ + t) * FEAT + k] : 0.0f;
      sh_x[idx] = (_Float16)v;
    }
    if (t + 1 < SEQ)  // uniform branch; lowers to global_prefetch_b8
      __builtin_prefetch(&x[((size_t)(bbase + (tid >> 5)) * SEQ + t + 1) * FEAT], 0, 0);
    __syncthreads();

    // ---- Layer 0 gates: g = x_t @ Wih0^T + h0 @ Whh0^T ----
    {
      v16h ax  = ldsA(sh_x, 32, 0);
      v16h ah0 = ldsA(sh_h0, HIDDEN, 0);
      v16h ah1 = ldsA(sh_h0, HIDDEN, 1);
      v8f acc0 = {}; v8f acc1 = {};
      acc0 = WMMA(ax,  Bih0a,  acc0);
      acc0 = WMMA(ah0, Bhh0a0, acc0);
      acc0 = WMMA(ah1, Bhh0a1, acc0);
      acc1 = WMMA(ax,  Bih0b,  acc1);
      acc1 = WMMA(ah0, Bhh0b0, acc1);
      acc1 = WMMA(ah1, Bhh0b1, acc1);
#pragma unroll
      for (int r = 0; r < 8; ++r) {
        sh_g[(cmb + r) * GATES + n0 + cn] = acc0[r];
        sh_g[(cmb + r) * GATES + n1 + cn] = acc1[r];
      }
    }
    __syncthreads();

    // ---- Layer 0 elementwise (i,f,g,o), c in registers ----
#pragma unroll
    for (int p = 0; p < 4; ++p) {
      int m = p * 4 + emb;
      const float* g = sh_g + m * GATES;
      float gi = g[en]        + sh_b0[en];
      float gf = g[64  + en]  + sh_b0[64  + en];
      float gz = g[128 + en]  + sh_b0[128 + en];
      float go = g[192 + en]  + sh_b0[192 + en];
      float cc = fsig(gf) * c0[p] + fsig(gi) * ftanh(gz);
      c0[p] = cc;
      sh_h0[m * HIDDEN + en] = (_Float16)(fsig(go) * ftanh(cc));
    }
    __syncthreads();

    // ---- Layer 1 gates: g = h0_new @ Wih1^T + h1 @ Whh1^T ----
    {
      v16h ai0 = ldsA(sh_h0, HIDDEN, 0);
      v16h ai1 = ldsA(sh_h0, HIDDEN, 1);
      v16h ah0 = ldsA(sh_h1, HIDDEN, 0);
      v16h ah1 = ldsA(sh_h1, HIDDEN, 1);
      v8f acc0 = {}; v8f acc1 = {};
      acc0 = WMMA(ai0, Bih1a0, acc0);
      acc0 = WMMA(ai1, Bih1a1, acc0);
      acc0 = WMMA(ah0, Bhh1a0, acc0);
      acc0 = WMMA(ah1, Bhh1a1, acc0);
      acc1 = WMMA(ai0, Bih1b0, acc1);
      acc1 = WMMA(ai1, Bih1b1, acc1);
      acc1 = WMMA(ah0, Bhh1b0, acc1);
      acc1 = WMMA(ah1, Bhh1b1, acc1);
#pragma unroll
      for (int r = 0; r < 8; ++r) {
        sh_g[(cmb + r) * GATES + n0 + cn] = acc0[r];
        sh_g[(cmb + r) * GATES + n1 + cn] = acc1[r];
      }
    }
    __syncthreads();

    // ---- Layer 1 elementwise ----
#pragma unroll
    for (int p = 0; p < 4; ++p) {
      int m = p * 4 + emb;
      const float* g = sh_g + m * GATES;
      float gi = g[en]        + sh_b1[en];
      float gf = g[64  + en]  + sh_b1[64  + en];
      float gz = g[128 + en]  + sh_b1[128 + en];
      float go = g[192 + en]  + sh_b1[192 + en];
      float cc = fsig(gf) * c1[p] + fsig(gi) * ftanh(gz);
      c1[p] = cc;
      sh_h1[m * HIDDEN + en] = (_Float16)(fsig(go) * ftanh(cc));
    }
    __syncthreads();
  }

  // Final FC on last h1 (after all WMMA; divergence is safe here).
  if (tid < MTILE * NCLS) {
    int m = tid >> 1, cls = tid & 1;
    float s = bfc[cls];
#pragma unroll 8
    for (int n = 0; n < HIDDEN; ++n)
      s += (float)sh_h1[m * HIDDEN + n] * Wfc[cls * HIDDEN + n];
    out[(bbase + m) * NCLS + cls] = s;
  }
}

extern "C" void kernel_launch(void* const* d_in, const int* in_sizes, int n_in,
                              void* d_out, int out_size, void* d_ws, size_t ws_size,
                              hipStream_t stream) {
  const float* x    = (const float*)d_in[0];
  const float* Wih0 = (const float*)d_in[1];
  const float* Whh0 = (const float*)d_in[2];
  const float* bih0 = (const float*)d_in[3];
  const float* bhh0 = (const float*)d_in[4];
  const float* Wih1 = (const float*)d_in[5];
  const float* Whh1 = (const float*)d_in[6];
  const float* bih1 = (const float*)d_in[7];
  const float* bhh1 = (const float*)d_in[8];
  const float* Wfc  = (const float*)d_in[9];
  const float* bfc  = (const float*)d_in[10];
  float* out = (float*)d_out;

  dim3 grid(BATCH / MTILE);   // 512 workgroups of 16 batch rows
  dim3 block(256);            // 8 wave32 waves; 2 N-tiles (32 gates) per wave
  hipLaunchKernelGGL(lstm2_fused_wmma, grid, block, 0, stream,
                     x, Wih0, Whh0, bih0, bhh0, Wih1, Whh1, bih1, bhh1,
                     Wfc, bfc, out);
}
